// ANINNP_43817256354319
// MI455X (gfx1250) — matrix operand order repaired and checked by reading, not compiled, not run on hardware
//
#include <hip/hip_runtime.h>

typedef _Float16 v16h __attribute__((ext_vector_type(16)));
typedef float    v8f  __attribute__((ext_vector_type(8)));

#define KP1 1312   // 1281 padded to multiple of 32
#define NATOT 1024 // B*A = 32*32

#if defined(__has_builtin)
#if __has_builtin(__builtin_amdgcn_global_load_async_to_lds_b64) && \
    __has_builtin(__builtin_amdgcn_s_wait_asynccnt)
#define USE_ASYNC_LDS 1
#endif
#endif

// builtin signature (from hipcc diagnostic): (v2i AS1*, v2i AS3*, imm int, imm int)
typedef int async_v2i __attribute__((vector_size(8)));
#define GLOBAL_AS64(p) ((__attribute__((address_space(1))) async_v2i*)(p))
#define LDS_AS64(p)    ((__attribute__((address_space(3))) async_v2i*)(p))

__device__ __forceinline__ float celu01(float v) {
    return v > 0.f ? v : 0.1f * (expf(10.f * v) - 1.f);
}

// ---------------- prep: transpose + fp16-convert weights -------------------
__global__ void prep_kernel(const float* __restrict__ W1, const float* __restrict__ W2,
                            _Float16* __restrict__ W1T, _Float16* __restrict__ W2T) {
    int idx = blockIdx.x * 256 + threadIdx.x;
    const int NW1 = 64 * KP1;
    if (idx < NW1) {
        int n = idx / KP1, k = idx % KP1;
        W1T[idx] = (k < 1281) ? (_Float16)W1[k * 64 + n] : (_Float16)0.f;
    } else if (idx < NW1 + 32 * 64) {
        int r = idx - NW1;
        int n = r / 64, k = r % 64;
        W2T[r] = (_Float16)W2[k * 32 + n];
    }
}

// ---------------- AEV: one wave32 per atom ---------------------------------
__global__ void aev_kernel(const int* __restrict__ species, const float* __restrict__ coords,
                           const int* __restrict__ belong, const float* __restrict__ approx,
                           _Float16* __restrict__ X) {
    int atom = blockIdx.x;            // 0..1023
    int b = atom >> 5, i = atom & 31;
    int lane = threadIdx.x;           // 0..31

    __shared__ float sux[32], suy[32], suz[32], sd[32], sfca[32], sfcr[32];
    __shared__ int   ssp[32];
    __shared__ float srad[128];       // [8 species][16 shifts]
    __shared__ float sang[36 * 32];   // [36 pairs][(a,z) = lane]

    // neighbor setup: lane = j
    {
        int j = lane;
        float cjx = coords[(b * 32 + j) * 3 + 0];
        float cjy = coords[(b * 32 + j) * 3 + 1];
        float cjz = coords[(b * 32 + j) * 3 + 2];
        float cix = coords[(b * 32 + i) * 3 + 0];
        float ciy = coords[(b * 32 + i) * 3 + 1];
        float ciz = coords[(b * 32 + i) * 3 + 2];
        float dx = cjx - cix, dy = cjy - ciy, dz = cjz - ciz;
        float d2 = dx * dx + dy * dy + dz * dz;
        float d = (d2 > 0.f) ? sqrtf(d2) : 1.f;
        bool self = (j == i);
        float fcr = (!self && d <= 5.2f) ? (0.5f * cosf(3.14159265358979f * d / 5.2f) + 0.5f) : 0.f;
        float fca = (!self && d <= 3.5f) ? (0.5f * cosf(3.14159265358979f * d / 3.5f) + 0.5f) : 0.f;
        sd[j] = d; sfcr[j] = fcr; sfca[j] = fca;
        float inv = 1.f / d;
        sux[j] = dx * inv; suy[j] = dy * inv; suz[j] = dz * inv;
        ssp[j] = species[b * 32 + j] + (belong[b * 32 + j] - 1) * 4;
    }
    for (int q = lane; q < 128; q += 32)  srad[q] = 0.f;
    for (int q = lane; q < 1152; q += 32) sang[q] = 0.f;
    __syncthreads();

    // radial: lane = shift index r (lanes 0..15)
    if (lane < 16) {
        float shf = 0.9f + 0.26875f * (float)lane;
        for (int jj = 0; jj < 32; jj++) {
            float fr = sfcr[jj];
            if (fr != 0.f) {
                float t = sd[jj] - shf;
                srad[ssp[jj] * 16 + lane] += 0.25f * expf(-16.f * t * t) * fr;
            }
        }
    }
    __syncthreads();

    // angular: lane = (a,z) bin; loop unordered pairs j<k (uniform LDS branches)
    {
        int az_a = lane >> 3, az_z = lane & 7;
        float shfa = 0.9f + 0.65f * (float)az_a;
        float zang = 0.19634954084936f + 0.39269908169872f * (float)az_z; // pi/16 + z*pi/8
        float cz = cosf(zang), sz = sinf(zang);
        for (int jj = 0; jj < 31; jj++) {
            float faj = sfca[jj];
            if (faj == 0.f) continue;
            float ujx = sux[jj], ujy = suy[jj], ujz = suz[jj], dj = sd[jj];
            int sj = ssp[jj];
            for (int kk = jj + 1; kk < 32; kk++) {
                float fak = sfca[kk];
                if (fak == 0.f) continue;
                float w  = faj * fak;
                float ct = ujx * sux[kk] + ujy * suy[kk] + ujz * suz[kk];
                ct = fminf(1.f, fmaxf(-1.f, ct)) * 0.95f;     // TorchANI stabilizer
                float st = sqrtf(1.f - ct * ct);              // sin(theta) >= 0
                float ravg = 0.5f * (dj + sd[kk]);
                float t2 = ravg - shfa;
                float f2 = expf(-8.f * t2 * t2);
                float base = 0.5f * (1.f + ct * cz + st * sz);
                float f1 = base;                               // base^32 by squaring
                f1 *= f1; f1 *= f1; f1 *= f1; f1 *= f1; f1 *= f1;
                int s0 = sj, t0 = ssp[kk];
                if (s0 > t0) { int tm = s0; s0 = t0; t0 = tm; }
                int p = s0 * 8 - (s0 * (s0 - 1)) / 2 + (t0 - s0);  // triu index
                sang[p * 32 + lane] += 2.f * w * f2 * f1;      // lane-unique addr: no race
            }
        }
    }
    __syncthreads();

    // emit fp16 feature row [1312] = 128 radial | 1152 angular | approx | pad
    _Float16* xr = X + (size_t)atom * KP1;
    for (int q = lane; q < 128; q += 32) xr[q] = (_Float16)srad[q];
    for (int p = 0; p < 36; p++) xr[128 + p * 32 + lane] = (_Float16)sang[p * 32 + lane];
    if (lane == 0) {
        xr[1280] = (_Float16)approx[atom];
        for (int q = 1281; q < KP1; q++) xr[q] = (_Float16)0.f;
    }
}

// ---------------- layer 1 GEMM: [1024,1312] x [1312,64], WMMA f16->f32 -----
#ifdef USE_ASYNC_LDS
// Async double-buffered A-tile staging: 4 waves share one 16x32 A tile per
// K-chunk, loaded via GLOBAL_LOAD_ASYNC_TO_LDS_B64 (ASYNCcnt pipeline).
__global__ void gemm1_kernel(const _Float16* __restrict__ X, const _Float16* __restrict__ W1T,
                             const float* __restrict__ b1, _Float16* __restrict__ H1) {
    __shared__ _Float16 sA[2][16][40];     // 40-half row stride: conflict-free frag reads
    int tid  = threadIdx.x;
    int lane = tid & 31, wv = tid >> 5;    // 4 waves -> 4 x 16-col tiles
    int l15  = lane & 15, hi = lane >> 4;
    int rowbase = blockIdx.x * 16;

    int lrow = tid >> 3, lseg = tid & 7;   // 128 threads cover 16 rows x 8 x 8B
    const _Float16* gsrc = X + (size_t)(rowbase + lrow) * KP1 + lseg * 4;
    _Float16* ldst[2] = { &sA[0][lrow][lseg * 4], &sA[1][lrow][lseg * 4] };
    const _Float16* wr = W1T + (size_t)(wv * 16 + l15) * KP1;

    __builtin_amdgcn_global_load_async_to_lds_b64(GLOBAL_AS64(gsrc), LDS_AS64(ldst[0]), 0, 0);

    v8f acc = {};
    const int NK = KP1 / 32;               // 41 K-chunks
    for (int kc = 0; kc < NK; kc++) {
        int k = kc * 32;
        if (kc + 1 < NK) {                 // prefetch next chunk, keep <=1 in flight
            __builtin_amdgcn_global_load_async_to_lds_b64(GLOBAL_AS64(gsrc + k + 32),
                                                          LDS_AS64(ldst[(kc + 1) & 1]), 0, 0);
            __builtin_amdgcn_s_wait_asynccnt(1);
        } else {
            __builtin_amdgcn_s_wait_asynccnt(0);
        }
        __syncthreads();                   // chunk kc visible to all waves
        const _Float16* as = &sA[kc & 1][l15][hi * 16];
        v16h a, bb;
        int bo = k + hi * 16;
#pragma unroll
        for (int e = 0; e < 8; e++) { a[e] = as[e]; a[e + 8] = as[8 + e]; }
#pragma unroll
        for (int e = 0; e < 16; e++) bb[e] = wr[bo + e];
        acc = __builtin_amdgcn_wmma_f32_16x16x32_f16(false, a, false, bb,
                                                     (short)0, acc, false, false);
        __syncthreads();                   // done reading before buffer reuse
    }
    int ncol = wv * 16 + l15;
    float bias = b1[ncol];
#pragma unroll
    for (int r = 0; r < 8; r++) {          // C/D: M = r + 8*hi, N = lane&15
        int m = rowbase + r + hi * 8;
        H1[m * 64 + ncol] = (_Float16)celu01(acc[r] + bias);
    }
}
#else
__global__ void gemm1_kernel(const _Float16* __restrict__ X, const _Float16* __restrict__ W1T,
                             const float* __restrict__ b1, _Float16* __restrict__ H1) {
    int lane = threadIdx.x & 31;
    int wv   = threadIdx.x >> 5;
    int l15  = lane & 15;
    int hi   = lane >> 4;
    int rowbase = blockIdx.x * 16;
    const _Float16* xr = X   + (size_t)(rowbase + l15) * KP1;
    const _Float16* wr = W1T + (size_t)(wv * 16 + l15) * KP1;
    v8f acc = {};
    for (int k = 0; k < KP1; k += 32) {
        __builtin_prefetch(xr + k + 128, 0, 3);   // WGP-scope prefetch
        v16h a, bb;
        int a1 = k + hi * 16, a2 = k + hi * 16 + 8;
        int bo = k + hi * 16;
#pragma unroll
        for (int e = 0; e < 8; e++) { a[e] = xr[a1 + e]; a[e + 8] = xr[a2 + e]; }
#pragma unroll
        for (int e = 0; e < 16; e++) bb[e] = wr[bo + e];
        acc = __builtin_amdgcn_wmma_f32_16x16x32_f16(false, a, false, bb,
                                                     (short)0, acc, false, false);
    }
    int ncol = wv * 16 + l15;
    float bias = b1[ncol];
#pragma unroll
    for (int r = 0; r < 8; r++) {
        int m = rowbase + r + hi * 8;
        H1[m * 64 + ncol] = (_Float16)celu01(acc[r] + bias);
    }
}
#endif

// ---------------- layer 2 GEMM: [1024,64] x [64,32], WMMA ------------------
__global__ void gemm2_kernel(const _Float16* __restrict__ H1, const _Float16* __restrict__ W2T,
                             const float* __restrict__ b2, float* __restrict__ H2) {
    int lane = threadIdx.x & 31;
    int wv   = threadIdx.x >> 5;          // 0..1
    int l15  = lane & 15;
    int hi   = lane >> 4;
    int rowbase = blockIdx.x * 16;
    const _Float16* xr = H1  + (size_t)(rowbase + l15) * 64;
    const _Float16* wr = W2T + (size_t)(wv * 16 + l15) * 64;
    v8f acc = {};
#pragma unroll
    for (int k = 0; k < 64; k += 32) {
        v16h a, bb;
        int a1 = k + hi * 16, a2 = k + hi * 16 + 8;
        int bo = k + hi * 16;
#pragma unroll
        for (int e = 0; e < 8; e++) { a[e] = xr[a1 + e]; a[e + 8] = xr[a2 + e]; }
#pragma unroll
        for (int e = 0; e < 16; e++) bb[e] = wr[bo + e];
        acc = __builtin_amdgcn_wmma_f32_16x16x32_f16(false, a, false, bb,
                                                     (short)0, acc, false, false);
    }
    int ncol = wv * 16 + l15;
    float bias = b2[ncol];
#pragma unroll
    for (int r = 0; r < 8; r++) {
        int m = rowbase + r + hi * 8;
        H2[m * 32 + ncol] = celu01(acc[r] + bias);
    }
}

// ---------------- tail: layers 3-4 + per-molecule reduction ----------------
__global__ void tail_kernel(const float* __restrict__ H2, const float* __restrict__ W3,
                            const float* __restrict__ b3, const float* __restrict__ W4,
                            const float* __restrict__ b4, float* __restrict__ out) {
    int b = blockIdx.x, a = threadIdx.x;          // one wave32 per molecule
    const float* h = H2 + (size_t)(b * 32 + a) * 32;
    float y = b4[0];
    for (int n = 0; n < 16; n++) {
        float s = b3[n];
#pragma unroll
        for (int k = 0; k < 32; k++) s += h[k] * W3[k * 16 + n];
        y += celu01(s) * W4[n];
    }
#pragma unroll
    for (int off = 16; off; off >>= 1) y += __shfl_xor(y, off, 32);
    if (a == 0) out[b] = y;
}

extern "C" void kernel_launch(void* const* d_in, const int* in_sizes, int n_in,
                              void* d_out, int out_size, void* d_ws, size_t ws_size,
                              hipStream_t stream) {
    const int*   species = (const int*)d_in[0];
    const float* coords  = (const float*)d_in[1];
    const int*   belong  = (const int*)d_in[2];
    const float* approx  = (const float*)d_in[3];
    const float* W1 = (const float*)d_in[4];
    const float* b1 = (const float*)d_in[5];
    const float* W2 = (const float*)d_in[6];
    const float* b2 = (const float*)d_in[7];
    const float* W3 = (const float*)d_in[8];
    const float* b3 = (const float*)d_in[9];
    const float* W4 = (const float*)d_in[10];
    const float* b4 = (const float*)d_in[11];
    float* out = (float*)d_out;

    // workspace carve-out (all regions fully overwritten every call)
    _Float16* X   = (_Float16*)d_ws;           // 1024*1312 fp16 feature matrix
    _Float16* W1T = X   + (size_t)NATOT * KP1; // 64*1312
    _Float16* W2T = W1T + (size_t)64 * KP1;    // 32*64
    _Float16* H1  = W2T + 32 * 64;             // 1024*64
    float*    H2  = (float*)(H1 + (size_t)NATOT * 64); // 1024*32 (offset is 16B aligned)

    int prep_threads = 64 * KP1 + 32 * 64;
    prep_kernel<<<(prep_threads + 255) / 256, 256, 0, stream>>>(W1, W2, W1T, W2T);
    aev_kernel<<<NATOT, 32, 0, stream>>>(species, coords, belong, approx, X);
    gemm1_kernel<<<NATOT / 16, 128, 0, stream>>>(X, W1T, b1, H1);
    gemm2_kernel<<<NATOT / 16, 64, 0, stream>>>(H1, W2T, b2, H2);
    tail_kernel<<<32, 32, 0, stream>>>(H2, W3, b3, W4, b4, out);
}
